// vitxxs_attention_35364760715889
// MI455X (gfx1250) — matrix operand order repaired and checked by reading, not compiled, not run on hardware
//
#include <hip/hip_runtime.h>
#include <hip/hip_bf16.h>

typedef __attribute__((ext_vector_type(16))) __bf16 v16bf;
typedef __attribute__((ext_vector_type(2)))  __bf16 v2bf;
typedef __attribute__((ext_vector_type(8)))  float  v8f;

#define BB 4
#define CC 64
#define NN 4096

// ---------------- WMMA + lane helpers ----------------

static __device__ __forceinline__ v8f wmma_bf16(v16bf a, v16bf b, v8f c) {
  // D = A(16x32 bf16) * B(32x16 bf16) + C(16x16 f32)
  return __builtin_amdgcn_wmma_f32_16x16x32_bf16(false, a, false, b, (short)0, c,
                                                 false, false);
}

union V16U { v16bf v; float4 f4[2]; };
union PBU  { v16bf v; unsigned u[8]; };

// A-fragment style load: 8 bf16 at p0, 8 bf16 at p1 (two b128 loads)
static __device__ __forceinline__ v16bf ld_pair(const __bf16* p0, const __bf16* p1) {
  V16U u;
  u.f4[0] = *(const float4*)p0;
  u.f4[1] = *(const float4*)p1;
  return u.v;
}
// 16 contiguous bf16
static __device__ __forceinline__ v16bf ld_cont(const __bf16* p) {
  return ld_pair(p, p + 8);
}

// SWAPX16: exchange lanes L <-> L+16 (group-of-32, xor=0x10, and=0x1f)
static __device__ __forceinline__ float swz_f(float x) {
  return __builtin_bit_cast(float,
      __builtin_amdgcn_ds_swizzle(__builtin_bit_cast(int, x), 0x401F));
}
static __device__ __forceinline__ unsigned swz_u(unsigned x) {
  return (unsigned)__builtin_amdgcn_ds_swizzle((int)x, 0x401F);
}
static __device__ __forceinline__ unsigned pack_bf(float a, float b) {
  v2bf t; t[0] = (__bf16)a; t[1] = (__bf16)b;
  return __builtin_bit_cast(unsigned, t);
}

// Generic LDS pointer -> raw LDS offset (low 32 bits of the generic address)
static __device__ __forceinline__ unsigned lds_off(const void* p) {
  return (unsigned)(uintptr_t)p;
}

// CDNA5 async global->LDS (ASYNCcnt-tracked, no VGPR staging).
// SADDR form: mem = SGPR_base + VGPR_i32_offset.
static __device__ __forceinline__ void async_ld_b128(unsigned lds_byte_off,
                                                     unsigned gl_byte_off,
                                                     const void* base) {
  asm volatile("global_load_async_to_lds_b128 %0, %1, %2"
               :
               : "v"(lds_byte_off), "v"(gl_byte_off), "s"(base)
               : "memory");
}

// ---------------- Projection kernel: Y = W X + b -------------
// p: 0=Q1 1=K1 2=V1 3=Q2 4=K2 5=V2
//   Q/K written transposed [n][c] bf16 (Q pre-scaled by 1/sqrt(C))
//   V  written [c][n] bf16

struct ProjArgs {
  const float* src[2];
  const float* w[6];
  const float* bias[6];
  __bf16* dst[6];
};

__global__ __launch_bounds__(128) void proj_kernel(ProjArgs args) {
  __shared__ __align__(16) __bf16 Wl[64][72];   // [o][c], padded
  __shared__ __align__(16) __bf16 Xt[64][72];   // [n_local][c], padded
  __shared__ float bs[64];

  const int t  = threadIdx.x;
  const int p  = blockIdx.y;
  const int b  = blockIdx.z;
  const int n0 = blockIdx.x * 64;

  const float* src = args.src[p < 3 ? 0 : 1];
  const float* w   = args.w[p];
  const float* bia = args.bias[p];
  __bf16* dst      = args.dst[p];
  const bool transposed = (p != 2 && p != 5);
  const float scale     = (p == 0 || p == 3) ? 0.125f : 1.0f; // 1/sqrt(64)

#pragma unroll
  for (int k = 0; k < 32; ++k) {          // weights -> LDS (bf16)
    int e = t + k * 128;
    Wl[e >> 6][e & 63] = (__bf16)w[e];
  }
#pragma unroll
  for (int k = 0; k < 32; ++k) {          // X block -> LDS transposed
    int e = t + k * 128;
    int c = e >> 6, j = e & 63;
    Xt[j][c] = (__bf16)src[((size_t)b * CC + c) * NN + n0 + j];
  }
  if (t < 64) bs[t] = bia[t];
  __syncthreads();

  const int wv  = t >> 5;          // wave -> o-tile
  const int lid = t & 15;
  const int hi  = (t >> 4) & 1;

  // A = W tile: lane row o, K=c {0..15}(lo) / {16..31}(hi)
  v16bf a0 = ld_cont(&Wl[wv * 16 + lid][hi * 16]);
  v16bf a1 = ld_cont(&Wl[wv * 16 + lid][32 + hi * 16]);

#pragma unroll
  for (int jt = 0; jt < 4; ++jt) {
    int nl = jt * 16 + lid;
    // B = X tile: lane col n, 16 contiguous c values
    v16bf b0 = ld_cont(&Xt[nl][hi * 16]);
    v16bf b1 = ld_cont(&Xt[nl][32 + hi * 16]);
    v8f acc = {};
    acc = wmma_bf16(a0, b0, acc);
    acc = wmma_bf16(a1, b1, acc);

    float vals[8];
#pragma unroll
    for (int r = 0; r < 8; ++r) {
      int o = wv * 16 + hi * 8 + r;
      vals[r] = (acc[r] + bs[o]) * scale;
    }
    const int n = n0 + jt * 16 + lid;
    if (transposed) {
#pragma unroll
      for (int r = 0; r < 8; r += 2) {
        int o = wv * 16 + hi * 8 + r;
        *(unsigned*)&dst[(size_t)(b * NN + n) * CC + o] = pack_bf(vals[r], vals[r + 1]);
      }
    } else {
#pragma unroll
      for (int r = 0; r < 8; ++r) {
        int o = wv * 16 + hi * 8 + r;
        dst[((size_t)b * CC + o) * NN + n] = (__bf16)vals[r];
      }
    }
  }
}

// ---------------- Fused bidirectional flash attention ----------------
// dir0: Q1 vs (K2,V2);  dir1: Q2 vs (K1,V1);  out = gamma*(O1+O2), [b][c][n]
// K/V tiles double-buffered in LDS, filled by async global->LDS loads.

__global__ __launch_bounds__(256) void xattn_kernel(
    const __bf16* __restrict__ Qt1, const __bf16* __restrict__ Kt1,
    const __bf16* __restrict__ V1,  const __bf16* __restrict__ Qt2,
    const __bf16* __restrict__ Kt2, const __bf16* __restrict__ V2,
    const float* __restrict__ gamma, float* __restrict__ out) {
  __shared__ __align__(16) __bf16 sK[2][2][32][72];  // [buf][dir][m][c], padded
  __shared__ __align__(16) __bf16 sV[2][2][64][40];  // [buf][dir][c][m], padded

  const int t   = threadIdx.x;
  const int b   = blockIdx.y;
  const int wv  = t >> 5;
  const int lid = t & 15;
  const int hi  = (t >> 4) & 1;
  const int qbase = blockIdx.x * 128 + wv * 16;  // 16 queries per wave
  const size_t mo = (size_t)b * NN * CC;

  const __bf16* Q[2] = { Qt1 + mo, Qt2 + mo };
  const __bf16* K[2] = { Kt2 + mo, Kt1 + mo };
  const __bf16* V[2] = { V2 + mo, V1 + mo };

  // Q B-fragments held in registers for the whole sweep
  v16bf qf[2][2];
#pragma unroll
  for (int d = 0; d < 2; ++d)
#pragma unroll
    for (int f = 0; f < 2; ++f)
      qf[d][f] = ld_pair(Q[d] + (size_t)(qbase + lid) * CC + f * 32 + hi * 8, Q[d] + (size_t)(qbase + lid) * CC + f * 32 + hi * 8 + 16);

  v8f oacc[2][4];
  const v8f zf = {};
#pragma unroll
  for (int d = 0; d < 2; ++d)
#pragma unroll
    for (int ct = 0; ct < 4; ++ct) oacc[d][ct] = zf;
  float mrun[2] = { -1e30f, -1e30f };
  float lrun[2] = { 0.f, 0.f };

  const int km = t >> 3, kc = (t & 7) * 8;   // K-tile copy coords (8 bf16 each)
  const int vc = t >> 2, vj = (t & 3) * 8;   // V-tile copy coords

  // Issue the 4 async b128 loads for one (buf, m0) tile set.
  auto stage = [&](int buf, int m0) {
#pragma unroll
    for (int d = 0; d < 2; ++d) {
      async_ld_b128(lds_off(&sK[buf][d][km][kc]),
                    (unsigned)(((m0 + km) * CC + kc) * 2), K[d]);
      async_ld_b128(lds_off(&sV[buf][d][vc][vj]),
                    (unsigned)((vc * NN + m0 + vj) * 2), V[d]);
    }
  };

  constexpr int NIT = NN / 32;
  stage(0, 0);  // prologue

  for (int it = 0; it < NIT; ++it) {
    const int buf = it & 1;
    if (it + 1 < NIT) {
      stage(buf ^ 1, (it + 1) * 32);                    // prefetch next tile
      asm volatile("s_wait_asynccnt 0x4" ::: "memory"); // current 4 loads done
    } else {
      asm volatile("s_wait_asynccnt 0x0" ::: "memory");
    }
    __syncthreads();

#pragma unroll
    for (int d = 0; d < 2; ++d) {
      // S^T tiles: lane = query column n, VGPR r = key row m
      v8f st[2];
#pragma unroll
      for (int mt = 0; mt < 2; ++mt) {
        int mr = mt * 16 + lid;
        v16bf k0 = ld_pair(&sK[buf][d][mr][hi * 8],      &sK[buf][d][mr][hi * 8 + 16]);
        v16bf k1 = ld_pair(&sK[buf][d][mr][32 + hi * 8], &sK[buf][d][mr][32 + hi * 8 + 16]);
        v8f acc = zf;
        acc = wmma_bf16(k0, qf[d][0], acc);
        acc = wmma_bf16(k1, qf[d][1], acc);
        st[mt] = acc;
      }

      // online softmax: per-lane reduce + one SWAPX16 to merge lane halves
      float mx = -1e30f;
#pragma unroll
      for (int r = 0; r < 8; ++r) mx = fmaxf(mx, fmaxf(st[0][r], st[1][r]));
      mx = fmaxf(mx, swz_f(mx));
      float mnew  = fmaxf(mrun[d], mx);
      float alpha = __expf(mrun[d] - mnew);
      mrun[d] = mnew;

      float p0[8], p1[8], rs = 0.f;
#pragma unroll
      for (int r = 0; r < 8; ++r) {
        p0[r] = __expf(st[0][r] - mnew);
        p1[r] = __expf(st[1][r] - mnew);
        rs += p0[r] + p1[r];
      }
      rs += swz_f(rs);
      lrun[d] = lrun[d] * alpha + rs;

      // P^T -> B-operand fragment (K=m): pack + half-lane swap
      PBU pbf;
#pragma unroll
      for (int v = 0; v < 4; ++v) {
        unsigned pa  = pack_bf(p0[2 * v], p0[2 * v + 1]);
        unsigned pb  = pack_bf(p1[2 * v], p1[2 * v + 1]);
        unsigned spa = swz_u(pa);
        unsigned spb = swz_u(pb);
        pbf.u[v]     = hi ? spb : pa;
        pbf.u[4 + v] = hi ? pb  : spa;
      }

      // O^T += V * P^T  (A = V rows c, contiguous m from LDS)
#pragma unroll
      for (int ct = 0; ct < 4; ++ct) {
        int cr = ct * 16 + lid;
#pragma unroll
        for (int r = 0; r < 8; ++r) oacc[d][ct][r] *= alpha;
        v16bf av = ld_cont(&sV[buf][d][cr][hi * 16]);
        oacc[d][ct] = wmma_bf16(av, pbf.v, oacc[d][ct]);
      }
    }
    __syncthreads();  // all waves done reading buf -> safe to overwrite next iter
  }

  const float g  = gamma[0];
  const float i0 = 1.0f / lrun[0];
  const float i1 = 1.0f / lrun[1];
  const int n = qbase + lid;
#pragma unroll
  for (int ct = 0; ct < 4; ++ct)
#pragma unroll
    for (int r = 0; r < 8; ++r) {
      int c = ct * 16 + hi * 8 + r;
      out[((size_t)b * CC + c) * NN + n] = g * (oacc[0][ct][r] * i0 + oacc[1][ct][r] * i1);
    }
}

// ---------------- host launch ----------------

extern "C" void kernel_launch(void* const* d_in, const int* in_sizes, int n_in,
                              void* d_out, int out_size, void* d_ws, size_t ws_size,
                              hipStream_t stream) {
  (void)in_sizes; (void)n_in; (void)out_size; (void)ws_size;

  ProjArgs pa;
  pa.src[0] = (const float*)d_in[0];   // f_global
  pa.src[1] = (const float*)d_in[1];   // f_modal
  for (int p = 0; p < 6; ++p) {
    pa.w[p]    = (const float*)d_in[2 + 2 * p];
    pa.bias[p] = (const float*)d_in[3 + 2 * p];
  }
  __bf16* wsb = (__bf16*)d_ws;
  const size_t msz = (size_t)BB * NN * CC;     // per-matrix bf16 elements (2 MB each)
  for (int p = 0; p < 6; ++p) pa.dst[p] = wsb + (size_t)p * msz;

  dim3 gp(NN / 64, 6, BB);
  proj_kernel<<<gp, 128, 0, stream>>>(pa);

  dim3 ga(NN / 128, BB);
  xattn_kernel<<<ga, 256, 0, stream>>>(pa.dst[0], pa.dst[1], pa.dst[2],
                                       pa.dst[3], pa.dst[4], pa.dst[5],
                                       (const float*)d_in[14], (float*)d_out);
}
